// FernBitWord_tabular_44779329028744
// MI455X (gfx1250) — matrix operand order repaired and checked by reading, not compiled, not run on hardware
//
#include <hip/hip_runtime.h>

typedef __attribute__((ext_vector_type(16))) _Float16 v16h;
typedef __attribute__((ext_vector_type(8)))  float    v8f;
typedef __attribute__((ext_vector_type(4)))  float    v4f;

#define NROWS   200000
#define MK      1024
#define DDIM    32
#define NTILES  (NROWS / 16)   // 12500
#define MKTILES (MK / 16)      // 64
#define WPB     8              // waves per block (wave32)

// ---------------------------------------------------------------------------
// Prep: softmax over alpha rows (1024 x 32), stored as f16 row-major (each
// row is one B column, k = d), plus a fused epilogue-constant table:
//   ctab[mk] = { cc, cs }  with  cs = 1/(pos-neg+1e-29),  cc = -(th+neg)*cs
// so the epilogue is  B = clamp( fma(temp, cs, cc), 0, 1 ).
// One wave32 per row; lane d holds element d.
// ---------------------------------------------------------------------------
__global__ __launch_bounds__(256) void fern_prep(
    const float* __restrict__ alpha, const float* __restrict__ th,
    const float* __restrict__ amb, _Float16* __restrict__ sah,
    float2* __restrict__ ctab)
{
    int wave = (int)((blockIdx.x * blockDim.x + threadIdx.x) >> 5);
    int lane = (int)(threadIdx.x & 31);
    if (wave >= MK) return;                    // wave-uniform

    float x = alpha[(size_t)wave * DDIM + lane];
    float m = x;
    #pragma unroll
    for (int off = 16; off > 0; off >>= 1)
        m = fmaxf(m, __shfl_xor(m, off, 32));
    float e = __expf(x - m);
    float s = e;
    #pragma unroll
    for (int off = 16; off > 0; off >>= 1)
        s += __shfl_xor(s, off, 32);
    sah[(size_t)wave * DDIM + lane] = (_Float16)(e / s);

    if (lane == 0) {
        float pos = amb[2 * wave + 0];
        float neg = amb[2 * wave + 1];
        float cs  = 1.0f / (pos - neg + 1e-29f);
        float cc  = -(th[wave] + neg) * cs;
        ctab[wave] = make_float2(cc, cs);
    }
}

// ---------------------------------------------------------------------------
// Main: each wave computes a 16(N) x 1024(MK) slab as 64 WMMA tiles.
// A fragment (T rows, f32 -> f16, NT loads) built once per wave; B fragments
// are two 8-half runs of the softmax table (L1/L2 resident); epilogue is
// select + fma + clamp; output streamed with non-temporal stores.
// ---------------------------------------------------------------------------
__global__ __launch_bounds__(256) void fern_main(
    const float* __restrict__ T, const _Float16* __restrict__ sah,
    const float2* __restrict__ ctab, float* __restrict__ out)
{
    int lane  = (int)(threadIdx.x & 31);
    int wave  = (int)(threadIdx.x >> 5);
    int ntile = (int)blockIdx.x * WPB + wave;
    if (ntile >= NTILES) return;               // wave-uniform

    int rlo  = lane & 15;                      // row within tile (A: M=lane%16)
    int hi   = lane >> 4;                      // half-wave select
    int base = hi * 8;                         // A K-base per ISA 16-bit layout

    // ---- A fragment: row = ntile*16 + rlo, halfs 0..7 = K base..base+7,
    //      halfs 8..15 = K 16+base .. 16+base+7 (ISA 05_wmma 16-bit A layout)
    const v4f* Trow = (const v4f*)(T + (size_t)(ntile * 16 + rlo) * DDIM);
    v4f t0 = __builtin_nontemporal_load(Trow + (base >> 2));
    v4f t1 = __builtin_nontemporal_load(Trow + (base >> 2) + 1);
    v4f t2 = __builtin_nontemporal_load(Trow + 4 + (base >> 2));
    v4f t3 = __builtin_nontemporal_load(Trow + 5 + (base >> 2));
    float tf[16] = { t0[0], t0[1], t0[2], t0[3],  t1[0], t1[1], t1[2], t1[3],
                     t2[0], t2[1], t2[2], t2[3],  t3[0], t3[1], t3[2], t3[3] };
    v16h a;
    #pragma unroll
    for (int i = 0; i < 16; ++i) a[i] = (_Float16)tf[i];

    int col = rlo;                             // B/D column = lane%16
    int kb  = hi * 16;                         // unused; B lanes>=16 hold K=8..15, 24..31
    size_t outRowBase = (size_t)(ntile * 16 + hi * 8) * MK;

    for (int t = 0; t < MKTILES; ++t) {
        int mk = t * 16 + col;

        // B fragment: halfs 0..7 = K 8*hi..8*hi+7, halfs 8..15 = K 16+8*hi..16+8*hi+7
        typedef __attribute__((ext_vector_type(8))) _Float16 v8h_;
        v8h_ b0 = *(const v8h_*)(sah + (size_t)mk * DDIM + base);
        v8h_ b1 = *(const v8h_*)(sah + (size_t)mk * DDIM + 16 + base);
        v16h b = __builtin_shufflevector(b0, b1, 0,1,2,3,4,5,6,7,8,9,10,11,12,13,14,15);

        v8f c = {};
        c = __builtin_amdgcn_wmma_f32_16x16x32_f16(
                /*neg_a=*/false, a, /*neg_b=*/false, b,
                /*c_mod=*/(short)0, c, /*reuse_a=*/false, /*reuse_b=*/false);

        float2 k2 = ctab[mk];                  // {cc, cs}

        // D layout: VGPR j -> row (j + 8*hi), lane%16 -> col
        #pragma unroll
        for (int j = 0; j < 8; ++j) {
            float v = c[j];
            v = (__builtin_fabsf(v) < 1e-5f) ? 0.0f : v;
            float r = fmaf(v, k2.y, k2.x);
            r = fminf(fmaxf(r, 0.0f), 1.0f);
            __builtin_nontemporal_store(r, out + outRowBase + (size_t)j * MK
                                              + (size_t)mk);
        }
    }
}

extern "C" void kernel_launch(void* const* d_in, const int* in_sizes, int n_in,
                              void* d_out, int out_size, void* d_ws, size_t ws_size,
                              hipStream_t stream) {
    (void)in_sizes; (void)n_in; (void)out_size; (void)ws_size;
    const float* T     = (const float*)d_in[0];
    const float* alpha = (const float*)d_in[1];
    const float* th    = (const float*)d_in[2];
    const float* amb   = (const float*)d_in[3];
    float* out = (float*)d_out;

    _Float16* sah  = (_Float16*)d_ws;                          // 64 KB
    float2*   ctab = (float2*)((char*)d_ws + MK * DDIM * sizeof(_Float16));

    fern_prep<<<(MK * 32 + 255) / 256, 256, 0, stream>>>(alpha, th, amb,
                                                         sah, ctab);
    fern_main<<<(NTILES + WPB - 1) / WPB, 256, 0, stream>>>(T, sah, ctab, out);
}
